// MPNN_17257178596039
// MI455X (gfx1250) — compile-verified, matrix operand order checked by reading, not run on hardware
//
#include <hip/hip_runtime.h>
#include <stdint.h>

// MPNN: out = relu(x@W_upd + segmean(adj, x@W_msg) + bias)
// B=8, N=2048, F=U=256. GEMMs via v_wmma_f32_16x16x32_f16;
// adjacency staged to LDS with global_load_async_to_lds_b128 (ASYNCcnt pipelined).

#define B_ 8
#define N_ 2048
#define F_ 256
#define U_ 256

typedef __attribute__((ext_vector_type(16))) _Float16 v16h;
typedef __attribute__((ext_vector_type(8)))  _Float16 v8h;
typedef __attribute__((ext_vector_type(8)))  float    v8f;
typedef __attribute__((ext_vector_type(4)))  float    v4f;

__device__ __forceinline__ v16h cat8(v8h lo, v8h hi) {
  v16h r;
#pragma unroll
  for (int j = 0; j < 8; ++j) { r[j] = lo[j]; r[8 + j] = hi[j]; }
  return r;
}

__device__ __forceinline__ v16h cvt_a(v4f a, v4f b, v4f c, v4f d) {
  v16h r;
#pragma unroll
  for (int j = 0; j < 4; ++j) {
    r[j]      = (_Float16)a[j];
    r[4 + j]  = (_Float16)b[j];
    r[8 + j]  = (_Float16)c[j];
    r[12 + j] = (_Float16)d[j];
  }
  return r;
}

// {0,1} int pair -> packed f16 pair. f16(1.0) = 0x3C00; no cross-half carry.
__device__ __forceinline__ uint32_t pack01(int a0, int a1) {
  return (uint32_t)(a0 | (a1 << 16)) * 0x3C00u;
}

// ---------------------------------------------------------------------------
// Kernel 0: W[f][u] f32 -> Wt[u][f] f16 (so B-fragments load contiguously)
// ---------------------------------------------------------------------------
__global__ void __launch_bounds__(256)
mpnn_wt_kernel(const float* __restrict__ W, _Float16* __restrict__ Wt) {
  int t = blockIdx.x * 256 + threadIdx.x;   // 65536 threads
  int u = t >> 8;
  int f = t & 255;
  Wt[u * F_ + f] = (_Float16)W[f * U_ + u];
}

// ---------------------------------------------------------------------------
// Kernel 1: scale[b][j] = indeg>0 ? 1/indeg : 0 ; also warms 192MB L2 with adj
// ---------------------------------------------------------------------------
__global__ void __launch_bounds__(256)
mpnn_deg_kernel(const int* __restrict__ adj, float* __restrict__ scale) {
  int t = blockIdx.x * 256 + threadIdx.x;   // 16384 threads
  int b = t >> 11;
  int j = t & (N_ - 1);
  const int* p = adj + (size_t)b * N_ * N_ + j;
  int d = 0;
  for (int i = 0; i < N_; ++i) d += p[(size_t)i * N_];
  scale[t] = (d > 0) ? (1.0f / (float)d) : 0.0f;
}

// ---------------------------------------------------------------------------
// Kernel 2: fused feature transform.
//   Ht[b][u][i]  = f16( x @ W_msg )   (transposed for contiguous B-frags later)
//   XWu[b][n][u] = f32( x @ W_upd )
// One wave per 16x16 tile; shared A-fragment feeds both WMMAs.
// ---------------------------------------------------------------------------
__global__ void __launch_bounds__(256)
mpnn_xform_kernel(const float* __restrict__ x,
                  const _Float16* __restrict__ WtMsg,
                  const _Float16* __restrict__ WtUpd,
                  _Float16* __restrict__ Ht,
                  float* __restrict__ XWu) {
  const int lane = threadIdx.x & 31;
  const int wave = threadIdx.x >> 5;
  const int gid  = blockIdx.x * 8 + wave;      // 0..16383
  const int b    = gid >> 11;
  const int rem  = gid & 2047;
  const int n0   = (rem >> 4) << 4;            // node-tile base (M)
  const int u0   = (rem & 15) << 4;            // u-tile base (N)

  const int m    = lane & 15;                  // A row == B col == D col
  const int hi   = lane >> 4;
  const int koff = hi << 3;                    // K sub-offset per lane half

  v8f acc_m = {};
  v8f acc_u = {};

  const float*    xrow = x     + (size_t)(b * N_ + n0 + m) * F_;
  const _Float16* wm   = WtMsg + (size_t)(u0 + m) * F_;
  const _Float16* wu   = WtUpd + (size_t)(u0 + m) * F_;

#pragma unroll 2
  for (int k0 = 0; k0 < F_; k0 += 32) {
    v4f x0 = *(const v4f*)(xrow + k0 + koff);
    v4f x1 = *(const v4f*)(xrow + k0 + koff + 4);
    v4f x2 = *(const v4f*)(xrow + k0 + 16 + koff);
    v4f x3 = *(const v4f*)(xrow + k0 + 16 + koff + 4);
    v16h a = cvt_a(x0, x1, x2, x3);

    v16h bm = cat8(*(const v8h*)(wm + k0 + koff),
                   *(const v8h*)(wm + k0 + 16 + koff));
    v16h bu = cat8(*(const v8h*)(wu + k0 + koff),
                   *(const v8h*)(wu + k0 + 16 + koff));

    acc_m = __builtin_amdgcn_wmma_f32_16x16x32_f16(
        false, a, false, bm, (short)0, acc_m, false, false);
    acc_u = __builtin_amdgcn_wmma_f32_16x16x32_f16(
        false, a, false, bu, (short)0, acc_u, false, false);
  }

  // D layout: VGPR r -> row (n0 + 8*hi + r), col (u0 + m)
  const int ibase = n0 + (hi << 3);
  const int u     = u0 + m;

  v8h hh;
#pragma unroll
  for (int r = 0; r < 8; ++r) hh[r] = (_Float16)acc_m[r];
  *(v8h*)(Ht + (size_t)(b * U_ + u) * N_ + ibase) = hh;

#pragma unroll
  for (int r = 0; r < 8; ++r)
    XWu[(size_t)(b * N_ + ibase + r) * U_ + u] = acc_u[r];
}

// ---------------------------------------------------------------------------
// Kernel 3: message GEMM (msg = adj^T @ H) + fused epilogue.
// Block = 16 waves handles one (b, j-tile); wave w owns u-tile w.
// adj staged to LDS with global_load_async_to_lds_b128, double-buffered 8KB
// slabs (128 K-rows x 16 j int32), pipelined on ASYNCcnt.
// ---------------------------------------------------------------------------
__global__ void __launch_bounds__(512)
mpnn_msg_kernel(const int* __restrict__ adj,
                const _Float16* __restrict__ Ht,
                const float* __restrict__ XWu,
                const float* __restrict__ scale,
                const float* __restrict__ bias,
                float* __restrict__ out) {
  __shared__ __align__(16) int rawA[2][128 * 16];   // [i][j] raw int32 slabs

  const int b    = blockIdx.x >> 7;
  const int j0   = (blockIdx.x & 127) << 4;
  const int lane = threadIdx.x & 31;
  const int wave = threadIdx.x >> 5;               // u-tile id
  const int u0   = wave << 4;
  const int m    = lane & 15;
  const int hi   = lane >> 4;
  const int koff = hi << 3;

  // staging coordinates: thread copies 16B = 4 consecutive j of one i-row
  const int ii = threadIdx.x >> 2;                 // 0..127
  const int jw = (threadIdx.x & 3) << 2;           // 0,4,8,12

  v8f acc = {};
  const _Float16* hrow = Ht + (size_t)(b * U_ + u0 + m) * N_;
  const int*      acol = adj + (size_t)b * N_ * N_ + j0;

  // issue slab 0 -> buf 0 (one async b128 per lane: 512B per wave)
  {
    const int* src = acol + (size_t)ii * N_ + jw;
    uint32_t dst = (uint32_t)(uintptr_t)(&rawA[0][ii * 16 + jw]);
    asm volatile("global_load_async_to_lds_b128 %0, %1, off"
                 :: "v"(dst), "v"(src) : "memory");
  }

  for (int s = 0; s < 16; ++s) {
    const int i0 = s << 7;                         // 128 K-rows per slab
    if (s < 15) {                                  // issue slab s+1
      const int* src = acol + (size_t)(i0 + 128 + ii) * N_ + jw;
      uint32_t dst = (uint32_t)(uintptr_t)(&rawA[(s + 1) & 1][ii * 16 + jw]);
      asm volatile("global_load_async_to_lds_b128 %0, %1, off"
                   :: "v"(dst), "v"(src) : "memory");
      asm volatile("s_wait_asynccnt 1" ::: "memory");   // slab s has landed
    } else {
      asm volatile("s_wait_asynccnt 0" ::: "memory");
    }
    __syncthreads();                               // cross-wave visibility

    const int* rp = &rawA[s & 1][0];
#pragma unroll
    for (int ks = 0; ks < 4; ++ks) {
      const int kb = (ks << 5) + koff;
      union { uint32_t u[8]; v16h h; } au;
#pragma unroll
      for (int r = 0; r < 4; ++r) {
        const int k = kb + 2 * r;
        au.u[r]     = pack01(rp[k * 16 + m],        rp[(k + 1) * 16 + m]);
        au.u[4 + r] = pack01(rp[(k + 16) * 16 + m], rp[(k + 17) * 16 + m]);
      }
      v16h bb = cat8(*(const v8h*)(hrow + i0 + (ks << 5) + koff),
                     *(const v8h*)(hrow + i0 + (ks << 5) + 16 + koff));
      acc = __builtin_amdgcn_wmma_f32_16x16x32_f16(
          false, au.h, false, bb, (short)0, acc, false, false);
    }
    __syncthreads();                               // buf reuse protection
  }

  const float bn    = bias[u0 + m];
  const int   jbase = j0 + (hi << 3);
#pragma unroll
  for (int r = 0; r < 8; ++r) {
    float s  = scale[b * N_ + jbase + r];
    size_t o = (size_t)(b * N_ + jbase + r) * U_ + u0 + m;
    float v  = acc[r] * s + XWu[o] + bn;
    out[o]   = v > 0.0f ? v : 0.0f;
  }
}

// ---------------------------------------------------------------------------
extern "C" void kernel_launch(void* const* d_in, const int* in_sizes, int n_in,
                              void* d_out, int out_size, void* d_ws, size_t ws_size,
                              hipStream_t stream) {
  const float* x    = (const float*)d_in[0];
  const int*   adj  = (const int*)d_in[1];
  const float* Wmsg = (const float*)d_in[2];
  const float* Wupd = (const float*)d_in[3];
  const float* bias = (const float*)d_in[4];
  float*       out  = (float*)d_out;

  // Workspace layout (bytes):
  //   [0, 128K)          WtMsg  f16 [U][F]
  //   [128K, 256K)       WtUpd  f16 [U][F]
  //   [256K, 320K)       scale  f32 [B][N]
  //   [320K, 320K+8M)    Ht     f16 [B][U][N]
  //   [.., +16M)         XWu    f32 [B][N][U]
  char* ws = (char*)d_ws;
  _Float16* WtMsg = (_Float16*)(ws);
  _Float16* WtUpd = (_Float16*)(ws + (size_t)131072);
  float*    scl   = (float*)   (ws + (size_t)262144);
  _Float16* Ht    = (_Float16*)(ws + (size_t)327680);
  float*    XWu   = (float*)   (ws + (size_t)327680 + (size_t)8388608);

  mpnn_wt_kernel<<<256, 256, 0, stream>>>(Wmsg, WtMsg);
  mpnn_wt_kernel<<<256, 256, 0, stream>>>(Wupd, WtUpd);
  mpnn_deg_kernel<<<64, 256, 0, stream>>>(adj, scl);
  mpnn_xform_kernel<<<2048, 256, 0, stream>>>(x, WtMsg, WtUpd, Ht, XWu);
  mpnn_msg_kernel<<<1024, 512, 0, stream>>>(adj, Ht, XWu, scl, bias, out);
}
